// GAU_33689723470189
// MI455X (gfx1250) — matrix operand order repaired and checked by reading, not compiled, not run on hardware
//
#include <hip/hip_runtime.h>

#define BB   4
#define SS   4096
#define DD   512
#define HH   1024
#define N2H  2048
#define QKD  128
#define NTOT 2176   // 2*HID + QK_DIM
#define OUTD 8

typedef _Float16 half_t;
typedef __attribute__((ext_vector_type(16))) _Float16 v16h;
typedef __attribute__((ext_vector_type(8)))  _Float16 v8h;
typedef __attribute__((ext_vector_type(8)))  float    v8f;

union Frag { v16h v; v8h h[2]; };

// A-operand fragment (16xK row-major): lo lanes hold K{k0..k0+7, k0+16..k0+23},
// hi lanes K{k0+8..15, k0+24..31}. Caller pre-adds (lane>=16 ? 8 : 0) halves.
__device__ __forceinline__ v16h load_afrag(const half_t* p) {
    Frag f; f.h[0] = *(const v8h*)(p); f.h[1] = *(const v8h*)(p + 16); return f.v;
}
// B-operand fragment (KxN, stored N-major so K is contiguous per lane):
// lo lanes hold K{k0..k0+7, k0+16..k0+23}, hi lanes K{k0+8..15, k0+24..31}. Caller pre-adds (lane>=16 ? 8 : 0).
__device__ __forceinline__ v16h load_bfrag(const half_t* p) {
    Frag f; f.h[0] = *(const v8h*)(p); f.h[1] = *(const v8h*)(p + 16); return f.v;
}

// silu with HW approximate reciprocal (v_rcp_f32) instead of IEEE div chain.
__device__ __forceinline__ float silu_f(float x) {
    return x * __builtin_amdgcn_rcpf(1.0f + __expf(-x));
}

// ---------------- Kernel 1: LayerNorm, f32 -> f16 ----------------
__global__ void ln_kernel(const float* __restrict__ x, const float* __restrict__ g,
                          const float* __restrict__ bt, half_t* __restrict__ nh) {
    int row  = (blockIdx.x * blockDim.x + threadIdx.x) >> 5;   // one wave per row
    int lane = threadIdx.x & 31;
    const float* xr = x + (size_t)row * DD;
    float vals[16], sum = 0.f, sq = 0.f;
    #pragma unroll
    for (int c = 0; c < 16; ++c) {
        float t = xr[c * 32 + lane];
        vals[c] = t; sum += t; sq += t * t;
    }
    #pragma unroll
    for (int m = 16; m >= 1; m >>= 1) {
        sum += __shfl_xor(sum, m, 32);
        sq  += __shfl_xor(sq,  m, 32);
    }
    float mu  = sum * (1.0f / DD);
    float var = sq * (1.0f / DD) - mu * mu;
    float rs  = __builtin_amdgcn_rsqf(var + 1e-5f);            // v_rsq_f32
    half_t* orow = nh + (size_t)row * DD;
    #pragma unroll
    for (int c = 0; c < 16; ++c) {
        int k = c * 32 + lane;
        orow[k] = (half_t)((vals[c] - mu) * rs * g[k] + bt[k]);
    }
}

// ------------- Kernel 0b: weights -> f16, transposed (N-major) -------------
__global__ void wconv_kernel(const float* __restrict__ Wh, const float* __restrict__ Wqk,
                             half_t* __restrict__ WT) {
    int idx = blockIdx.x * blockDim.x + threadIdx.x;
    if (idx >= NTOT * DD) return;
    int n = idx >> 9;          // / 512
    int k = idx & (DD - 1);
    float v = (n < N2H) ? Wh[(size_t)k * N2H + n] : Wqk[(size_t)k * QKD + (n - N2H)];
    WT[idx] = (half_t)v;
}

// ------- Kernel 2: fused projection GEMM, one 16x16 tile per wave ----------
// C[16384, 2176] = normed_f16 @ WT^T ; epilogue: silu + route to vT/gate/q/k
__global__ void proj_kernel(const half_t* __restrict__ nh, const half_t* __restrict__ WT,
                            const float* __restrict__ bh, const float* __restrict__ bqk,
                            const float* __restrict__ gamma, const float* __restrict__ beta,
                            half_t* __restrict__ vT, float* __restrict__ gate,
                            half_t* __restrict__ qo, half_t* __restrict__ ko) {
    int wid  = (blockIdx.x * blockDim.x + threadIdx.x) >> 5;
    int lane = threadIdx.x & 31;
    const int NT = NTOT / 16;                 // 136 N-tiles
    int nt = wid % NT, mt = wid / NT;
    int n0 = nt * 16, I0 = mt * 16;
    int nlo  = lane & 15;
    int hi8  = (lane >> 4) * 8;               // A/C fragment lane offset
    int hi16 = (lane >> 4) * 16;              // unused; B fragment also uses hi8

    const half_t* arow = nh + (size_t)(I0 + nlo) * DD + hi8;
    const half_t* brow = WT + (size_t)(n0 + nlo) * DD + hi8;

    v8f acc = {};
    #pragma unroll 4
    for (int kk = 0; kk < DD; kk += 32) {
        v16h a = load_afrag(arow + kk);
        v16h b = load_bfrag(brow + kk);
        acc = __builtin_amdgcn_wmma_f32_16x16x32_f16(false, a, false, b,
                                                     (short)0, acc, false, false);
    }
    int n = n0 + nlo;                         // output column (fixed per lane)
    float bias = (n < N2H) ? bh[n] : bqk[n - N2H];
    float z[8];
    #pragma unroll
    for (int r = 0; r < 8; ++r) z[r] = silu_f(acc[r] + bias);   // row M = hi8 + r

    if (n0 < HH) {
        // v half -> vT[b][h][s] (f16, s-contiguous => one b128 store per lane)
        int b  = I0 >> 12;
        int s0 = (I0 & (SS - 1)) + hi8;
        v8h p;
        #pragma unroll
        for (int r = 0; r < 8; ++r) p[r] = (half_t)z[r];
        *(v8h*)(vT + ((size_t)(b * HH + n) * SS + s0)) = p;
    } else if (n0 < N2H) {
        int h = n - HH;
        #pragma unroll
        for (int r = 0; r < 8; ++r)
            gate[(size_t)(I0 + hi8 + r) * HH + h] = z[r];
    } else {
        int d = n - N2H;
        float g0 = gamma[d], b0 = beta[d];
        float g1 = gamma[QKD + d], b1 = beta[QKD + d];
        #pragma unroll
        for (int r = 0; r < 8; ++r) {
            size_t idx = (size_t)(I0 + hi8 + r) * QKD + d;
            qo[idx] = (half_t)(z[r] * g0 + b0);
            ko[idx] = (half_t)(z[r] * g1 + b1);
        }
    }
}

// -------- Kernel 3: fused relu^2 attention, flash-style, WMMA ------------
// Block = 8 waves, one 16-row q tile, full H=1024 accumulated in registers.
__global__ void __launch_bounds__(256) attn_kernel(const half_t* __restrict__ q,
                                                   const half_t* __restrict__ k,
                                                   const half_t* __restrict__ vT,
                                                   float* __restrict__ Vacc) {
    __shared__ half_t P[16 * 136];            // 16 x 128 relu^2 tile, padded stride
    const int PST = 136;
    int w    = threadIdx.x >> 5;              // wave id 0..7
    int lane = threadIdx.x & 31;
    int nlo  = lane & 15;
    int hi8  = (lane >> 4) * 8;
    int hi16 = (lane >> 4) * 16;

    int b  = blockIdx.x >> 8;                 // batch
    int I0 = (blockIdx.x & 255) * 16;         // row tile within batch

    // Resident q fragments: 16 rows x 128 (4 A-fragments)
    const half_t* qrow = q + ((size_t)b * SS + I0 + nlo) * QKD + hi8;
    v16h qf[4];
    #pragma unroll
    for (int f = 0; f < 4; ++f) qf[f] = load_afrag(qrow + f * 32);

    v8f zero = {};
    v8f acc[8];
    #pragma unroll
    for (int t = 0; t < 8; ++t) acc[t] = zero;

    const float scale = 1.0f / (float)SS;
    const half_t* kb = k  + (size_t)b * SS * QKD;
    const half_t* vb = vT + (size_t)b * HH * SS;

    for (int jc = 0; jc < SS / 128; ++jc) {
        // --- this wave's 16x16 S tile at columns j0..j0+15 ---
        int j0 = jc * 128 + w * 16;
        const half_t* krow = kb + (size_t)(j0 + nlo) * QKD + hi8;
        v8f s = zero;
        #pragma unroll
        for (int f = 0; f < 4; ++f) {
            v16h bf = load_bfrag(krow + f * 32);
            s = __builtin_amdgcn_wmma_f32_16x16x32_f16(false, qf[f], false, bf,
                                                       (short)0, s, false, false);
        }
        // prefetch next chunk's k tile into cache while AV math runs
        if (jc + 1 < SS / 128)
            __builtin_prefetch(krow + 128 * QKD, 0, 0);   // global_prefetch_b8
        // relu(s/seq)^2 -> f16 P tile in LDS (row M, col = w*16+nlo)
        #pragma unroll
        for (int r = 0; r < 8; ++r) {
            float t = s[r] * scale;
            t = t > 0.0f ? t : 0.0f;
            P[(hi8 + r) * PST + (w * 16 + nlo)] = (half_t)(t * t);
        }
        __syncthreads();
        // --- accumulate P(16x128) @ v(128 x this wave's 128 H cols) ---
        #pragma unroll
        for (int kk = 0; kk < 4; ++kk) {
            const half_t* pp = &P[nlo * PST + kk * 32 + hi8];
            Frag pa;
            pa.h[0] = *(const v8h*)(pp);
            pa.h[1] = *(const v8h*)(pp + 16);
            #pragma unroll
            for (int t = 0; t < 8; ++t) {
                const half_t* vrow = vb + (size_t)(w * 128 + t * 16 + nlo) * SS
                                        + jc * 128 + kk * 32 + hi8;
                v16h bf = load_bfrag(vrow);
                acc[t] = __builtin_amdgcn_wmma_f32_16x16x32_f16(false, pa.v, false, bf,
                                                                (short)0, acc[t], false, false);
            }
        }
        __syncthreads();
    }
    // epilogue: Vacc[b*S + I0 + M][h] (f32)
    #pragma unroll
    for (int t = 0; t < 8; ++t) {
        int h = w * 128 + t * 16 + nlo;
        #pragma unroll
        for (int r = 0; r < 8; ++r)
            Vacc[((size_t)b * SS + I0 + hi8 + r) * HH + h] = acc[t][r];
    }
}

// -------- Kernel 4: out = (Vacc * gate) @ Wo + bo  (N=8, VALU + shfl) -----
__global__ void out_kernel(const float* __restrict__ Vacc, const float* __restrict__ gate,
                           const float* __restrict__ Wo, const float* __restrict__ bo,
                           float* __restrict__ out) {
    int row  = (blockIdx.x * blockDim.x + threadIdx.x) >> 5;   // one wave per row
    int lane = threadIdx.x & 31;
    const float* vr = Vacc + (size_t)row * HH;
    const float* gr = gate + (size_t)row * HH;
    float a[8] = {0, 0, 0, 0, 0, 0, 0, 0};
    for (int c = 0; c < HH / 32; ++c) {
        int kk = c * 32 + lane;
        float t = vr[kk] * gr[kk];
        const float* wrow = Wo + (size_t)kk * OUTD;
        #pragma unroll
        for (int n = 0; n < 8; ++n) a[n] += t * wrow[n];
    }
    #pragma unroll
    for (int n = 0; n < 8; ++n) {
        #pragma unroll
        for (int m = 16; m >= 1; m >>= 1) a[n] += __shfl_xor(a[n], m, 32);
    }
    if (lane == 0) {
        #pragma unroll
        for (int n = 0; n < 8; ++n) out[(size_t)row * OUTD + n] = a[n] + bo[n];
    }
}

extern "C" void kernel_launch(void* const* d_in, const int* in_sizes, int n_in,
                              void* d_out, int out_size, void* d_ws, size_t ws_size,
                              hipStream_t stream) {
    (void)in_sizes; (void)n_in; (void)out_size; (void)ws_size;
    const float* x     = (const float*)d_in[0];
    const float* ln_g  = (const float*)d_in[1];
    const float* ln_b  = (const float*)d_in[2];
    const float* Wh    = (const float*)d_in[3];
    const float* bh    = (const float*)d_in[4];
    const float* Wqk   = (const float*)d_in[5];
    const float* bqk   = (const float*)d_in[6];
    const float* gamma = (const float*)d_in[7];
    const float* beta  = (const float*)d_in[8];
    const float* Wo    = (const float*)d_in[9];
    const float* bo    = (const float*)d_in[10];
    float* out = (float*)d_out;

    char* p = (char*)d_ws;
    half_t* nh   = (half_t*)p;  p += (size_t)BB * SS * DD * 2;     // 16.8 MB
    half_t* WT   = (half_t*)p;  p += (size_t)NTOT * DD * 2;        //  2.2 MB
    half_t* qb   = (half_t*)p;  p += (size_t)BB * SS * QKD * 2;    //  4.2 MB
    half_t* kb   = (half_t*)p;  p += (size_t)BB * SS * QKD * 2;    //  4.2 MB
    half_t* vT   = (half_t*)p;  p += (size_t)BB * HH * SS * 2;     // 33.6 MB
    float*  gate = (float*)p;   p += (size_t)BB * SS * HH * 4;     // 67.1 MB
    float*  Vacc = (float*)p;   p += (size_t)BB * SS * HH * 4;     // 67.1 MB

    ln_kernel   <<<(BB * SS) / 8, 256, 0, stream>>>(x, ln_g, ln_b, nh);
    wconv_kernel<<<(NTOT * DD + 255) / 256, 256, 0, stream>>>(Wh, Wqk, WT);
    proj_kernel <<<((BB * SS / 16) * (NTOT / 16)) / 8, 256, 0, stream>>>(
        nh, WT, bh, bqk, gamma, beta, vT, gate, qb, kb);
    attn_kernel <<<BB * (SS / 16), 256, 0, stream>>>(qb, kb, vT, Vacc);
    out_kernel  <<<(BB * SS) / 8, 256, 0, stream>>>(Vacc, gate, Wo, bo, out);
}
